// TwoToOneDNPU_16595753632207
// MI455X (gfx1250) — matrix operand order repaired and checked by reading, not compiled, not run on hardware
//
#include <hip/hip_runtime.h>

#define H   90
#define HP  96          // padded hidden width (6 x 16)
#define TPB 256         // 8 waves of 32
#define NW  8
#define RPB 128         // rows per block (16 per wave)

typedef __attribute__((ext_vector_type(16))) _Float16 v16h;
typedef __attribute__((ext_vector_type(8)))  float    v8f;

union HVec { v16h v; unsigned int u[8]; };

__device__ __forceinline__ float eluf(float v) {
    return v > 0.f ? v : (__expf(v) - 1.f);
}

// ---------------------------------------------------------------------------
// Stage A: x' = scale*x+offset ; x1 = MLP_in1(x') ; x2 = MLP_in2(x')
//          write c = [x1,x2], per-block partial sums of (S1,Q1,S2,Q2)
// ---------------------------------------------------------------------------
__global__ __launch_bounds__(TPB) void dnpu_stageA(
    const float* __restrict__ x,
    const float* __restrict__ offset,
    const float* __restrict__ scale,
    const float* __restrict__ W1a, const float* __restrict__ b1a,
    const float* __restrict__ W2a, const float* __restrict__ b2a,
    const float* __restrict__ W3a, const float* __restrict__ b3a,
    const float* __restrict__ W1b, const float* __restrict__ b1b,
    const float* __restrict__ W2b, const float* __restrict__ b2b,
    const float* __restrict__ W3b, const float* __restrict__ b3b,
    float* __restrict__ c_ws, float* __restrict__ partials, int N)
{
    __shared__ _Float16 sW2T[2][HP * HP];    // W2 transposed, f16, zero-padded
    __shared__ _Float16 sW3B[2][16 * HP];    // W3 as B-tile: col 0 = W3, transposed
    __shared__ float    sW1[2][2][HP];
    __shared__ float    sb1[2][HP];
    __shared__ float    sb2[2][HP];
    __shared__ float    sX[NW][16][2];
    __shared__ _Float16 sH[NW][16 * HP];     // h1 (f16) then reused for h2
    __shared__ float    sRed[NW][4];

    const int tid  = threadIdx.x;
    const int w    = tid >> 5;
    const int lane = tid & 31;
    const int hi   = lane >> 4;
    const int ln   = lane & 15;

    // ---- stage small params into LDS (zero-padded to HP) ----
    for (int i = tid; i < 2 * HP; i += TPB) {
        int m = i / HP, col = i % HP;
        const float* b1 = m ? b1b : b1a;
        const float* b2 = m ? b2b : b2a;
        const float* W1 = m ? W1b : W1a;
        float vb1 = 0.f, vb2 = 0.f, vw10 = 0.f, vw11 = 0.f;
        if (col < H) { vb1 = b1[col]; vb2 = b2[col];
                       vw10 = W1[col]; vw11 = W1[H + col]; }
        sb1[m][col] = vb1; sb2[m][col] = vb2;
        sW1[m][0][col] = vw10; sW1[m][1][col] = vw11;
    }
    // ---- W2 transposed into LDS f16: sW2T[m][n*HP+k] = W2[k,n] ----
    for (int i = tid; i < 2 * HP * HP; i += TPB) {
        int m = i / (HP * HP);
        int r = i % (HP * HP);
        int n = r / HP, k = r % HP;
        const float* W2 = m ? W2b : W2a;
        float v = (n < H && k < H) ? W2[k * H + n] : 0.f;
        sW2T[m][r] = (_Float16)v;
    }
    // ---- W3 as a 96x16 B-tile (transposed [n][k]), only column n==0 live ----
    for (int i = tid; i < 2 * 16 * HP; i += TPB) {
        int m = i / (16 * HP);
        int r = i % (16 * HP);
        int n = r / HP, k = r % HP;
        const float* W3 = m ? W3b : W3a;
        sW3B[m][r] = (_Float16)((n == 0 && k < H) ? W3[k] : 0.f);
    }

    const int rowBase = blockIdx.x * RPB + w * 16;
    const float sc = scale[0];
    const float off0 = offset[0], off1 = offset[1];
    const float vb3a = b3a[0], vb3b = b3b[0];

    if (lane < 16) {
        int row = rowBase + lane;
        float x0 = 0.f, x1v = 0.f;
        if (row < N) { x0 = x[2 * row]; x1v = x[2 * row + 1]; }
        sX[w][lane][0] = sc * x0 + off0;
        sX[w][lane][1] = sc * x1v + off1;
    }
    __syncthreads();

    float r3[2][8];

#pragma unroll
    for (int m = 0; m < 2; ++m) {
        // ---- layer 1: 2 -> 96, ELU, store f16 (per-wave tile 16xHP) ----
#pragma unroll
        for (int cc = 0; cc < 3; ++cc) {
            int col = lane + cc * 32;
            float w10 = sW1[m][0][col], w11 = sW1[m][1][col], bb = sb1[m][col];
            for (int r = 0; r < 16; ++r) {
                float v = sX[w][r][0] * w10 + sX[w][r][1] * w11 + bb;
                sH[w][r * HP + col] = (_Float16)eluf(v);
            }
        }
        __syncthreads();

        // ---- layer 2: WMMA  C[16x96] = h1[16x96] * W2[96x96] ----
        v8f C[6] = {};
#pragma unroll
        for (int t = 0; t < 3; ++t) {
            HVec A;
#pragma unroll
            for (int r = 0; r < 8; ++r) {
                int k = 32 * t + (r < 4 ? hi * 8 + 2 * r : 16 + hi * 8 + 2 * (r - 4));
                __builtin_memcpy(&A.u[r], &sH[w][ln * HP + k], 4);
            }
#pragma unroll
            for (int j = 0; j < 6; ++j) {
                HVec B;
#pragma unroll
                for (int r = 0; r < 8; ++r) {
                    int k = 32 * t + (r < 4 ? hi * 8 + 2 * r : 16 + hi * 8 + 2 * (r - 4));
                    __builtin_memcpy(&B.u[r], &sW2T[m][(16 * j + ln) * HP + k], 4);
                }
                C[j] = __builtin_amdgcn_wmma_f32_16x16x32_f16(
                           false, A.v, false, B.v, (short)0, C[j], false, false);
            }
        }
        __syncthreads();

        // ---- epilogue: h2 = ELU(C + b2) -> reuse sH (f16) ----
#pragma unroll
        for (int j = 0; j < 6; ++j) {
#pragma unroll
            for (int r = 0; r < 8; ++r) {
                int row = r + 8 * hi;
                int col = 16 * j + ln;
                sH[w][row * HP + col] = (_Float16)eluf(C[j][r] + sb2[m][col]);
            }
        }
        __syncthreads();

        // ---- layer 3 via WMMA: B-tile = W3 in column 0 (cols 1..15 zero) ----
        v8f C3 = {};
#pragma unroll
        for (int t = 0; t < 3; ++t) {
            HVec A;
#pragma unroll
            for (int r = 0; r < 8; ++r) {
                int k = 32 * t + (r < 4 ? hi * 8 + 2 * r : 16 + hi * 8 + 2 * (r - 4));
                __builtin_memcpy(&A.u[r], &sH[w][ln * HP + k], 4);
            }
            HVec B;
#pragma unroll
            for (int r = 0; r < 8; ++r) {
                int k = 32 * t + (r < 4 ? hi * 8 + 2 * r : 16 + hi * 8 + 2 * (r - 4));
                __builtin_memcpy(&B.u[r], &sW3B[m][ln * HP + k], 4);
            }
            C3 = __builtin_amdgcn_wmma_f32_16x16x32_f16(
                     false, A.v, false, B.v, (short)0, C3, false, false);
        }
        // C3[r] = dot(h2[row r+8*hi], W3) on lanes with ln==0; zero elsewhere
#pragma unroll
        for (int r = 0; r < 8; ++r) r3[m][r] = C3[r];
        __syncthreads();   // before next MLP overwrites sH
    }

    // ---- write c and deterministic partial sums (lanes 0 and 16 hold rows) --
    float s1 = 0.f, q1 = 0.f, s2 = 0.f, q2 = 0.f;
    if (ln == 0) {
#pragma unroll
        for (int r = 0; r < 8; ++r) {
            int row = rowBase + r + 8 * hi;
            if (row < N) {
                float v1 = r3[0][r] + vb3a;
                float v2 = r3[1][r] + vb3b;
                c_ws[2 * row]     = v1;
                c_ws[2 * row + 1] = v2;
                s1 += v1; q1 += v1 * v1;
                s2 += v2; q2 += v2 * v2;
            }
        }
    }
#pragma unroll
    for (int d = 16; d >= 1; d >>= 1) {
        s1 += __shfl_xor(s1, d, 32);
        q1 += __shfl_xor(q1, d, 32);
        s2 += __shfl_xor(s2, d, 32);
        q2 += __shfl_xor(q2, d, 32);
    }
    if (lane == 0) { sRed[w][0] = s1; sRed[w][1] = q1; sRed[w][2] = s2; sRed[w][3] = q2; }
    __syncthreads();
    if (tid == 0) {
        float a0 = 0.f, a1 = 0.f, a2 = 0.f, a3 = 0.f;
        for (int i = 0; i < NW; ++i) {
            a0 += sRed[i][0]; a1 += sRed[i][1]; a2 += sRed[i][2]; a3 += sRed[i][3];
        }
        partials[(size_t)blockIdx.x * 4 + 0] = a0;
        partials[(size_t)blockIdx.x * 4 + 1] = a1;
        partials[(size_t)blockIdx.x * 4 + 2] = a2;
        partials[(size_t)blockIdx.x * 4 + 3] = a3;
    }
}

// ---------------------------------------------------------------------------
// Stage B: reduce partials -> batchnorm constants (deterministic tree)
// ---------------------------------------------------------------------------
__global__ __launch_bounds__(256) void dnpu_stageB(
    const float* __restrict__ partials, int nBlocks, int N,
    const float* __restrict__ running_var,
    const float* __restrict__ conversion_offset,
    float* __restrict__ consts)
{
    __shared__ float sh[256][4];
    float a[4] = {0.f, 0.f, 0.f, 0.f};
    for (int i = threadIdx.x; i < nBlocks; i += 256)
        for (int j = 0; j < 4; ++j) a[j] += partials[(size_t)i * 4 + j];
    for (int j = 0; j < 4; ++j) sh[threadIdx.x][j] = a[j];
    __syncthreads();
    for (int s = 128; s >= 1; s >>= 1) {
        if (threadIdx.x < s)
            for (int j = 0; j < 4; ++j) sh[threadIdx.x][j] += sh[threadIdx.x + s][j];
        __syncthreads();
    }
    if (threadIdx.x == 0) {
        float S1 = sh[0][0], Q1 = sh[0][1], S2 = sh[0][2], Q2 = sh[0][3];
        float invN = 1.f / (float)N;
        float m1 = S1 * invN, m2 = S2 * invN;
        float v1 = fmaxf(Q1 * invN - m1 * m1, 0.f);
        float v2 = fmaxf(Q2 * invN - m2 * m2, 0.f);
        float is1 = rsqrtf(v1 + 1e-5f), is2 = rsqrtf(v2 + 1e-5f);
        float f = (float)N / (float)(N - 1);
        float rv1 = 0.9f * running_var[0] + 0.1f * v1 * f;
        float rv2 = 0.9f * running_var[1] + 0.1f * v2 * f;
        float std1 = sqrtf(0.5f * (rv1 + rv2));
        consts[0] = m1;  consts[1] = m2;
        consts[2] = is1; consts[3] = is2;
        consts[4] = 1.8f / (4.f * std1);   // post-clip scale
        consts[5] = 2.f * std1;            // cut
        consts[6] = conversion_offset[0];
    }
}

// ---------------------------------------------------------------------------
// Stage C: normalize/clip c, run output MLP, write out[N] (fp32)
// ---------------------------------------------------------------------------
__global__ __launch_bounds__(TPB) void dnpu_stageC(
    const float* __restrict__ c_ws, const float* __restrict__ consts,
    const float* __restrict__ W1, const float* __restrict__ b1,
    const float* __restrict__ W2, const float* __restrict__ b2,
    const float* __restrict__ W3, const float* __restrict__ b3,
    float* __restrict__ out, int N)
{
    __shared__ _Float16 sW2T[HP * HP];
    __shared__ _Float16 sW3B[16 * HP];
    __shared__ float    sW1s[2][HP];
    __shared__ float    sb1s[HP];
    __shared__ float    sb2s[HP];
    __shared__ float    sX[NW][16][2];
    __shared__ _Float16 sH[NW][16 * HP];

    const int tid  = threadIdx.x;
    const int w    = tid >> 5;
    const int lane = tid & 31;
    const int hi   = lane >> 4;
    const int ln   = lane & 15;

    for (int col = tid; col < HP; col += TPB) {
        float vb1 = 0.f, vb2 = 0.f, vw10 = 0.f, vw11 = 0.f;
        if (col < H) { vb1 = b1[col]; vb2 = b2[col];
                       vw10 = W1[col]; vw11 = W1[H + col]; }
        sb1s[col] = vb1; sb2s[col] = vb2;
        sW1s[0][col] = vw10; sW1s[1][col] = vw11;
    }
    for (int r = tid; r < HP * HP; r += TPB) {
        int n = r / HP, k = r % HP;
        sW2T[r] = (_Float16)((n < H && k < H) ? W2[k * H + n] : 0.f);
    }
    for (int r = tid; r < 16 * HP; r += TPB) {
        int n = r / HP, k = r % HP;
        sW3B[r] = (_Float16)((n == 0 && k < H) ? W3[k] : 0.f);
    }

    const float m1 = consts[0], m2 = consts[1];
    const float is1 = consts[2], is2 = consts[3];
    const float aa = consts[4], cut = consts[5], co = consts[6];
    const float vb3 = b3[0];

    const int rowBase = blockIdx.x * RPB + w * 16;
    if (lane < 16) {
        int row = rowBase + lane;
        float c1 = 0.f, c2 = 0.f;
        if (row < N) { c1 = c_ws[2 * row]; c2 = c_ws[2 * row + 1]; }
        float h1 = aa * fminf(fmaxf((c1 - m1) * is1, -cut), cut) + co;
        float h2 = aa * fminf(fmaxf((c2 - m2) * is2, -cut), cut) + co;
        sX[w][lane][0] = h1;
        sX[w][lane][1] = h2;
    }
    __syncthreads();

    // layer 1
#pragma unroll
    for (int cc = 0; cc < 3; ++cc) {
        int col = lane + cc * 32;
        float w10 = sW1s[0][col], w11 = sW1s[1][col], bb = sb1s[col];
        for (int r = 0; r < 16; ++r) {
            float v = sX[w][r][0] * w10 + sX[w][r][1] * w11 + bb;
            sH[w][r * HP + col] = (_Float16)eluf(v);
        }
    }
    __syncthreads();

    // layer 2: WMMA
    v8f C[6] = {};
#pragma unroll
    for (int t = 0; t < 3; ++t) {
        HVec A;
#pragma unroll
        for (int r = 0; r < 8; ++r) {
            int k = 32 * t + (r < 4 ? hi * 8 + 2 * r : 16 + hi * 8 + 2 * (r - 4));
            __builtin_memcpy(&A.u[r], &sH[w][ln * HP + k], 4);
        }
#pragma unroll
        for (int j = 0; j < 6; ++j) {
            HVec B;
#pragma unroll
            for (int r = 0; r < 8; ++r) {
                int k = 32 * t + (r < 4 ? hi * 8 + 2 * r : 16 + hi * 8 + 2 * (r - 4));
                __builtin_memcpy(&B.u[r], &sW2T[(16 * j + ln) * HP + k], 4);
            }
            C[j] = __builtin_amdgcn_wmma_f32_16x16x32_f16(
                       false, A.v, false, B.v, (short)0, C[j], false, false);
        }
    }
    __syncthreads();

#pragma unroll
    for (int j = 0; j < 6; ++j) {
#pragma unroll
        for (int r = 0; r < 8; ++r) {
            int row = r + 8 * hi;
            int col = 16 * j + ln;
            sH[w][row * HP + col] = (_Float16)eluf(C[j][r] + sb2s[col]);
        }
    }
    __syncthreads();

    // layer 3 via WMMA
    v8f C3 = {};
#pragma unroll
    for (int t = 0; t < 3; ++t) {
        HVec A;
#pragma unroll
        for (int r = 0; r < 8; ++r) {
            int k = 32 * t + (r < 4 ? hi * 8 + 2 * r : 16 + hi * 8 + 2 * (r - 4));
            __builtin_memcpy(&A.u[r], &sH[w][ln * HP + k], 4);
        }
        HVec B;
#pragma unroll
        for (int r = 0; r < 8; ++r) {
            int k = 32 * t + (r < 4 ? hi * 8 + 2 * r : 16 + hi * 8 + 2 * (r - 4));
            __builtin_memcpy(&B.u[r], &sW3B[ln * HP + k], 4);
        }
        C3 = __builtin_amdgcn_wmma_f32_16x16x32_f16(
                 false, A.v, false, B.v, (short)0, C3, false, false);
    }
    if (ln == 0) {
#pragma unroll
        for (int r = 0; r < 8; ++r) {
            int row = rowBase + r + 8 * hi;
            if (row < N) out[row] = C3[r] + vb3;
        }
    }
}

// ---------------------------------------------------------------------------
extern "C" void kernel_launch(void* const* d_in, const int* in_sizes, int n_in,
                              void* d_out, int out_size, void* d_ws, size_t ws_size,
                              hipStream_t stream)
{
    const float* x        = (const float*)d_in[0];
    const float* offset   = (const float*)d_in[1];
    const float* scale    = (const float*)d_in[2];
    const float* conv_off = (const float*)d_in[3];
    const float* run_var  = (const float*)d_in[4];
    const float* in1_W1 = (const float*)d_in[5];
    const float* in1_b1 = (const float*)d_in[6];
    const float* in1_W2 = (const float*)d_in[7];
    const float* in1_b2 = (const float*)d_in[8];
    const float* in1_W3 = (const float*)d_in[9];
    const float* in1_b3 = (const float*)d_in[10];
    const float* in2_W1 = (const float*)d_in[11];
    const float* in2_b1 = (const float*)d_in[12];
    const float* in2_W2 = (const float*)d_in[13];
    const float* in2_b2 = (const float*)d_in[14];
    const float* in2_W3 = (const float*)d_in[15];
    const float* in2_b3 = (const float*)d_in[16];
    const float* out_W1 = (const float*)d_in[17];
    const float* out_b1 = (const float*)d_in[18];
    const float* out_W2 = (const float*)d_in[19];
    const float* out_b2 = (const float*)d_in[20];
    const float* out_W3 = (const float*)d_in[21];
    const float* out_b3 = (const float*)d_in[22];

    const int N = in_sizes[0] / 2;
    const int nBlocks = (N + RPB - 1) / RPB;

    float* ws       = (float*)d_ws;
    float* c_ws     = ws;                               // 2*N floats
    float* partials = ws + (size_t)2 * N;               // 4*nBlocks floats
    float* consts   = partials + (size_t)4 * nBlocks;   // 8 floats

    dnpu_stageA<<<nBlocks, TPB, 0, stream>>>(
        x, offset, scale,
        in1_W1, in1_b1, in1_W2, in1_b2, in1_W3, in1_b3,
        in2_W1, in2_b1, in2_W2, in2_b2, in2_W3, in2_b3,
        c_ws, partials, N);

    dnpu_stageB<<<1, 256, 0, stream>>>(partials, nBlocks, N, run_var, conv_off, consts);

    dnpu_stageC<<<nBlocks, TPB, 0, stream>>>(
        c_ws, consts,
        out_W1, out_b1, out_W2, out_b2, out_W3, out_b3,
        (float*)d_out, N);
}
